// HeteroGAT_11553462026819
// MI455X (gfx1250) — matrix operand order repaired and checked by reading, not compiled, not run on hardware
//
#include <hip/hip_runtime.h>
#include <math.h>

#define R   4
#define NP  100000
#define NF  256
#define HD  256
#define NC  40
#define NTOT (R * NP)          // 400000 nodes == edges

// ---------------------------------------------------------------- WMMA types
typedef __attribute__((ext_vector_type(16))) __bf16 v16bf;
typedef __attribute__((ext_vector_type(8)))  float  v8f;

union BFVec { unsigned short u[16]; v16bf v; };

__device__ __forceinline__ unsigned short f2bf(float f) {
  __bf16 b = (__bf16)f;                       // native RNE f32->bf16 convert
  return __builtin_bit_cast(unsigned short, b);
}

// order-preserving float<->uint encoding for atomic max over signed floats
__device__ __forceinline__ unsigned fenc(float f) {
  unsigned u = __float_as_uint(f);
  return (u & 0x80000000u) ? ~u : (u | 0x80000000u);
}
__device__ __forceinline__ float fdec(unsigned k) {
  unsigned u = (k & 0x80000000u) ? (k ^ 0x80000000u) : ~k;
  return __uint_as_float(u);
}
__device__ __forceinline__ float lrelu(float e) { return e > 0.f ? e : 0.2f * e; }

// ---------------------------------------------------------------- K1: h = x @ W (per relation), bf16 WMMA
// Block = 8 waves = 128 rows. Full-K A fragment kept in registers (64 VGPRs/lane);
// 4 phases loop over 64-col W strips staged in LDS (32KB). x is read exactly once.
#define BM 128
#define BN 64
#define KP 272   // LDS k-pitch in bf16 elems: 272*2=544B rows (32B aligned, bank-skewed)

__global__ __launch_bounds__(256)
void gat_gemm_h(const float* __restrict__ x, const float* __restrict__ W,
                float* __restrict__ h) {
  __shared__ __align__(64) unsigned short Wlds[BN * KP];

  const int r      = blockIdx.y;
  const int mblock = blockIdx.x * BM;
  const int tid    = threadIdx.x;
  const int lane   = tid & 31;
  const int wave   = tid >> 5;
  const int m0     = mblock + wave * 16;

  // A row for this lane; clamp out-of-range rows to 0 (their D rows are never stored,
  // and A row m only contributes to D row m) -> no divergence in the hot path.
  int rowA = m0 + (lane & 15);
  if (rowA >= NP) rowA = 0;
  const float* xrow = x + ((size_t)r * NP + rowA) * (size_t)NF;

  // ---- load the full K=256 A fragment into registers as bf16 (8 x 8 VGPRs).
  // A 16x32 layout (matches B): lane L holds row L&15,
  // K = kb + (L>>4)*16 + [0,16) contiguous in elems 0..15.
  BFVec afr[8];
  #pragma unroll
  for (int kb8 = 0; kb8 < 8; ++kb8) {
    const int k1 = kb8 * 32 + ((lane >> 4) << 4);
    const float4* p0 = reinterpret_cast<const float4*>(xrow + k1);
    const float4* p1 = reinterpret_cast<const float4*>(xrow + k1 + 8);
    float4 f0 = p0[0], f1 = p0[1], f2 = p1[0], f3 = p1[1];
    afr[kb8].u[0]  = f2bf(f0.x); afr[kb8].u[1]  = f2bf(f0.y);
    afr[kb8].u[2]  = f2bf(f0.z); afr[kb8].u[3]  = f2bf(f0.w);
    afr[kb8].u[4]  = f2bf(f1.x); afr[kb8].u[5]  = f2bf(f1.y);
    afr[kb8].u[6]  = f2bf(f1.z); afr[kb8].u[7]  = f2bf(f1.w);
    afr[kb8].u[8]  = f2bf(f2.x); afr[kb8].u[9]  = f2bf(f2.y);
    afr[kb8].u[10] = f2bf(f2.z); afr[kb8].u[11] = f2bf(f2.w);
    afr[kb8].u[12] = f2bf(f3.x); afr[kb8].u[13] = f2bf(f3.y);
    afr[kb8].u[14] = f2bf(f3.z); afr[kb8].u[15] = f2bf(f3.w);
  }

  const float* Wr = W + (size_t)r * NF * HD;

  for (int p = 0; p < 4; ++p) {
    const int n0 = p * BN;

    __syncthreads();   // previous phase's Wlds readers are done
    {
      // stage W[r][k][n0+nl] transposed into LDS as bf16: Wlds[nl*KP + k]
      const int nl  = tid & 63;
      const int kk0 = tid >> 6;                 // 0..3
      #pragma unroll 4
      for (int i = 0; i < 64; ++i) {
        int k = kk0 + (i << 2);                 // covers k = 0..255
        Wlds[nl * KP + k] = f2bf(Wr[k * HD + n0 + nl]);
      }
    }
    __syncthreads();

    v8f acc[4] = {};
    #pragma unroll
    for (int kb8 = 0; kb8 < 8; ++kb8) {
      // B fragment: lane L holds col L&15, K = kb + (L>>4)*16 + [0,16) contiguous
      const int ks = kb8 * 32 + ((lane >> 4) << 4);
      #pragma unroll
      for (int nt = 0; nt < 4; ++nt) {
        const int nl = nt * 16 + (lane & 15);
        v16bf b = *reinterpret_cast<const v16bf*>(&Wlds[nl * KP + ks]);
        acc[nt] = __builtin_amdgcn_wmma_f32_16x16x32_bf16(
            false, afr[kb8].v, false, b, (short)0, acc[nt], false, false);
      }
    }

    // store D (16x16 f32 layout: VGPR v -> row v + 8*(lane>>4), col lane&15)
    #pragma unroll
    for (int nt = 0; nt < 4; ++nt) {
      #pragma unroll
      for (int v = 0; v < 8; ++v) {
        const int row = m0 + v + ((lane >> 4) << 3);
        if (row < NP)
          h[((size_t)r * NP + row) * HD + n0 + nt * 16 + (lane & 15)] = acc[nt][v];
      }
    }
  }
}

// ---------------------------------------------------------------- K2: per-node attention scores (+ self-loop seeds segment max)
__global__ __launch_bounds__(256)
void gat_node_scores(const float* __restrict__ h, const float* __restrict__ att_src,
                     const float* __restrict__ att_dst, float* __restrict__ as_,
                     float* __restrict__ ad_, unsigned int* __restrict__ mkey) {
  const int wid  = (int)((blockIdx.x * blockDim.x + threadIdx.x) >> 5);
  const int lane = threadIdx.x & 31;
  if (wid >= NTOT) return;
  const int r = wid / NP;
  const float* hrow = h + (size_t)wid * HD;
  const float* av   = att_src + r * HD;
  const float* dv   = att_dst + r * HD;
  float s = 0.f, d = 0.f;
  #pragma unroll
  for (int j = 0; j < HD / 32; ++j) {
    float hv = hrow[lane + 32 * j];
    s += hv * av[lane + 32 * j];
    d += hv * dv[lane + 32 * j];
  }
  #pragma unroll
  for (int off = 16; off > 0; off >>= 1) {
    s += __shfl_xor(s, off);
    d += __shfl_xor(d, off);
  }
  if (lane == 0) {
    as_[wid] = s;
    ad_[wid] = d;
    mkey[wid] = fenc(lrelu(s + d));   // self-loop edge initializes the max
  }
}

// ---------------------------------------------------------------- K3: segment max over real edges
__global__ __launch_bounds__(256)
void gat_edge_max(const int* __restrict__ ei, const float* __restrict__ as_,
                  const float* __restrict__ ad_, unsigned int* __restrict__ mkey) {
  const int j = blockIdx.x * blockDim.x + threadIdx.x;
  if (j >= NTOT) return;
  const int r  = j / NP;
  const int gs = r * NP + ei[j];
  const int gd = r * NP + ei[NTOT + j];
  atomicMax(&mkey[gd], fenc(lrelu(as_[gs] + ad_[gd])));
}

// ---------------------------------------------------------------- K4a/K4b: softmax denominator
__global__ __launch_bounds__(256)
void gat_denom_init(const float* __restrict__ as_, const float* __restrict__ ad_,
                    const unsigned int* __restrict__ mkey, float* __restrict__ denom) {
  const int i = blockIdx.x * blockDim.x + threadIdx.x;
  if (i >= NTOT) return;
  denom[i] = __expf(lrelu(as_[i] + ad_[i]) - fdec(mkey[i]));   // self-loop term
}

__global__ __launch_bounds__(256)
void gat_edge_denom(const int* __restrict__ ei, const float* __restrict__ as_,
                    const float* __restrict__ ad_, const unsigned int* __restrict__ mkey,
                    float* __restrict__ denom) {
  const int j = blockIdx.x * blockDim.x + threadIdx.x;
  if (j >= NTOT) return;
  const int r  = j / NP;
  const int gs = r * NP + ei[j];
  const int gd = r * NP + ei[NTOT + j];
  atomicAdd(&denom[gd], __expf(lrelu(as_[gs] + ad_[gd]) - fdec(mkey[gd])));
}

// ---------------------------------------------------------------- K5a: out = bias + alpha_self * h
__global__ __launch_bounds__(256)
void gat_out_init(const float* __restrict__ h, const float* __restrict__ as_,
                  const float* __restrict__ ad_, const unsigned int* __restrict__ mkey,
                  const float* __restrict__ denom, const float* __restrict__ bias,
                  float* __restrict__ out) {
  const int i = blockIdx.x;
  const int c = threadIdx.x;
  const int r = i / NP;
  const float alpha = __expf(lrelu(as_[i] + ad_[i]) - fdec(mkey[i])) / (denom[i] + 1e-16f);
  out[(size_t)i * HD + c] = bias[r * HD + c] + alpha * h[(size_t)i * HD + c];
}

// ---------------------------------------------------------------- K5b: scatter-add alpha * h[src] into out[dst]
__global__ __launch_bounds__(256)
void gat_edge_aggr(const int* __restrict__ ei, const float* __restrict__ h,
                   const float* __restrict__ as_, const float* __restrict__ ad_,
                   const unsigned int* __restrict__ mkey, const float* __restrict__ denom,
                   float* __restrict__ out) {
  const int wid  = (int)((blockIdx.x * blockDim.x + threadIdx.x) >> 5);
  const int lane = threadIdx.x & 31;
  if (wid >= NTOT) return;
  const int r  = wid / NP;
  const int gs = r * NP + ei[wid];
  const int gd = r * NP + ei[NTOT + wid];
  const float alpha = __expf(lrelu(as_[gs] + ad_[gd]) - fdec(mkey[gd])) / (denom[gd] + 1e-16f);
  const float* hs = h + (size_t)gs * HD;
  float* od = out + (size_t)gd * HD;
  #pragma unroll
  for (int j = 0; j < HD / 32; ++j)
    atomicAdd(&od[lane + 32 * j], alpha * hs[lane + 32 * j]);
}

// ---------------------------------------------------------------- K5c: ELU
__global__ __launch_bounds__(256)
void elu_kernel(float* __restrict__ out, unsigned long long n) {
  unsigned long long i = blockIdx.x * (unsigned long long)blockDim.x + threadIdx.x;
  if (i < n) {
    float v = out[i];
    out[i] = v > 0.f ? v : (__expf(v) - 1.f);
  }
}

// ---------------------------------------------------------------- K6: final linear + log_softmax (wave per node)
__global__ __launch_bounds__(128)
void final_linear(const float* __restrict__ xf, const float* __restrict__ lw,
                  const float* __restrict__ lbias, float* __restrict__ out) {
  const int lane = threadIdx.x & 31;
  const int i = blockIdx.x * 4 + (threadIdx.x >> 5);
  if (i >= NP) return;

  // x_cat[i, k] = xf[(k>>8)*NP*HD + i*HD + (k&255)], k in [0,1024)
  float xr[32];
  #pragma unroll
  for (int j = 0; j < 32; ++j) {
    int k = lane + 32 * j;
    xr[j] = xf[((size_t)(k >> 8) * NP + i) * HD + (k & 255)];
  }

  float logitA = 0.f, logitB = 0.f;   // classes lane, lane+32
  for (int c = 0; c < NC; ++c) {
    const float* wrow = lw + c * (R * HD);
    float p = 0.f;
    #pragma unroll
    for (int j = 0; j < 32; ++j) p += xr[j] * wrow[lane + 32 * j];
    #pragma unroll
    for (int off = 16; off > 0; off >>= 1) p += __shfl_xor(p, off);
    p += lbias[c];
    if (c < 32) { if (lane == c)      logitA = p; }
    else        { if (lane == c - 32) logitB = p; }
  }

  float mx = logitA;
  if (lane < NC - 32) mx = fmaxf(mx, logitB);
  #pragma unroll
  for (int off = 16; off > 0; off >>= 1) mx = fmaxf(mx, __shfl_xor(mx, off));
  float se = __expf(logitA - mx) + ((lane < NC - 32) ? __expf(logitB - mx) : 0.f);
  #pragma unroll
  for (int off = 16; off > 0; off >>= 1) se += __shfl_xor(se, off);
  const float lse = mx + __logf(se);

  out[(size_t)i * NC + lane] = logitA - lse;
  if (lane < NC - 32) out[(size_t)i * NC + 32 + lane] = logitB - lse;
}

// ---------------------------------------------------------------- host launcher
extern "C" void kernel_launch(void* const* d_in, const int* in_sizes, int n_in,
                              void* d_out, int out_size, void* d_ws, size_t ws_size,
                              hipStream_t stream) {
  const float* x    = (const float*)d_in[0];
  const int*   ei   = (const int*)d_in[1];
  // d_in[2] = edge_type: implicit (relation blocks are contiguous), unused
  const float* W    = (const float*)d_in[3];
  const float* asv  = (const float*)d_in[4];
  const float* adv  = (const float*)d_in[5];
  const float* bias = (const float*)d_in[6];
  const float* lw   = (const float*)d_in[7];
  const float* lb   = (const float*)d_in[8];
  float* out = (float*)d_out;

  // workspace carve-up (fully rewritten before every read -> poison/replay safe)
  float* ws    = (float*)d_ws;
  float* h     = ws;                                   // NTOT*HD
  float* xbuf  = h + (size_t)NTOT * HD;                // NTOT*HD
  float* as_   = xbuf + (size_t)NTOT * HD;             // NTOT
  float* ad_   = as_ + NTOT;                           // NTOT
  unsigned int* mkey = (unsigned int*)(ad_ + NTOT);    // NTOT
  float* denom = (float*)(mkey + NTOT);                // NTOT

  const dim3 gGemm((NP + BM - 1) / BM, R);
  const int nodeBlocks = (NTOT + 255) / 256;
  const int waveBlocks = (NTOT * 32 + 255) / 256;      // wave-per-node/edge kernels

  const float* xcur = x;
  for (int layer = 0; layer < 2; ++layer) {
    gat_gemm_h<<<gGemm, 256, 0, stream>>>(xcur, W, h);
    gat_node_scores<<<waveBlocks, 256, 0, stream>>>(h, asv, adv, as_, ad_, mkey);
    gat_edge_max<<<nodeBlocks, 256, 0, stream>>>(ei, as_, ad_, mkey);
    gat_denom_init<<<nodeBlocks, 256, 0, stream>>>(as_, ad_, mkey, denom);
    gat_edge_denom<<<nodeBlocks, 256, 0, stream>>>(ei, as_, ad_, mkey, denom);
    gat_out_init<<<NTOT, HD, 0, stream>>>(h, as_, ad_, mkey, denom, bias, xbuf);
    gat_edge_aggr<<<waveBlocks, 256, 0, stream>>>(ei, h, as_, ad_, mkey, denom, xbuf);
    elu_kernel<<<(unsigned)(((unsigned long long)NTOT * HD + 255) / 256), 256, 0, stream>>>(
        xbuf, (unsigned long long)NTOT * HD);
    xcur = xbuf;   // layer-2 output safely overwrites its own input (dead after GEMM)
  }
  final_linear<<<(NP + 3) / 4, 128, 0, stream>>>(xbuf, lw, lb, out);
}
